// PFNet_63247688401093
// MI455X (gfx1250) — matrix operand order repaired and checked by reading, not compiled, not run on hardware
//
#include <hip/hip_runtime.h>
#include <stdint.h>

// ---------------------------------------------------------------- types
typedef __attribute__((ext_vector_type(16))) __bf16 v16bf;
typedef __attribute__((ext_vector_type(8)))  float  v8f;

union Frag {
  v16bf v;
  unsigned short u[16];
  unsigned int   d[8];
};

__device__ __forceinline__ unsigned short f2bf(float f) {
  unsigned int u = __float_as_uint(f);
  u += 0x7FFFu + ((u >> 16) & 1u);          // round-to-nearest-even
  return (unsigned short)(u >> 16);
}
__device__ __forceinline__ float bf2f(unsigned short h) {
  return __uint_as_float(((unsigned int)h) << 16);
}
__device__ __forceinline__ float selu_f(float x) {
  const float a = 1.6732632423543772f, s = 1.0507009873554805f;
  return s * (x > 0.f ? x : a * (expf(x) - 1.f));
}

// CDNA5 async global->LDS copy (VGPR-bypassing, tracked by ASYNCcnt).
// LDS address = low 32 bits of the generic pointer (flat-LDS truncation).
__device__ __forceinline__ void async_copy16(unsigned ldsOff, const void* g) {
  asm volatile("global_load_async_to_lds_b128 %0, %1, off"
               :: "v"(ldsOff), "v"((unsigned long long)(size_t)g) : "memory");
}
__device__ __forceinline__ void async_wait0() {
  asm volatile("s_wait_asynccnt 0" ::: "memory");
}

#define BM 128
#define BN 64
#define BK 32

// Packed-B layout ("fragment friendly"): for K-panel p (32 K values),
// dword[(p*Npad + n)*16 + kk/2] = (B[p*32+kk][n], B[p*32+kk+1][n]).
// A lane's 16 K values for column n then sit in two 4-dword runs starting 8 dwords apart.

// ------------------------------------------------ generic bf16 WMMA GEMM
// C[bz*M + m, n] = act( A[bz] (M x Kpad bf16, row-major) @ B[bz] (packed) + bias )
// 8 waves: 4x2, each wave computes a 32x32 patch (2x2 v_wmma_f32_16x16x32_bf16).
__global__ __launch_bounds__(256) void k_gemm(
    const unsigned short* __restrict__ A, int lda, long long sA,
    const unsigned short* __restrict__ Bp, long long sB,
    const float* __restrict__ bias,
    float* __restrict__ Cf, int ldc,
    unsigned short* __restrict__ Cb, int ldcb,
    int M, int Npad, int Kpad, int Nreal, int act)
{
  const int bz = blockIdx.z;
  A  += (long long)bz * sA;
  Bp += (long long)bz * sB;
  const int n0 = blockIdx.x * BN;
  const int m0 = blockIdx.y * BM;
  const int tid  = threadIdx.x;
  const int lane = tid & 31;
  const int wave = tid >> 5;
  const int waveM = wave >> 1;     // 0..3 (32 rows each)
  const int waveN = wave & 1;      // 0..1 (32 cols each)

  __shared__ uint4 As4[512];       // 128 x 32 bf16, row-major [m][16 dw]
  __shared__ uint4 Bs4[256];       // 64 cols x 16 dw (packed-B panel slice)
  unsigned int* As32 = (unsigned int*)As4;
  unsigned int* Bs32 = (unsigned int*)Bs4;
  const unsigned int* A32 = (const unsigned int*)A;
  const uint4* B4 = (const uint4*)Bp;
  const int ldaDw = lda >> 1;

  const bool fullB = (n0 + BN) <= Npad;             // block-uniform
  const long long aOff0 = (long long)(m0 + (tid >> 2)) * ldaDw + ((tid & 3) << 2);
  const long long aOff1 = aOff0 + 64LL * ldaDw;
  const unsigned asOff0 = (unsigned)(size_t)&As4[tid];
  const unsigned asOff1 = (unsigned)(size_t)&As4[tid + 256];
  const unsigned bsOff  = (unsigned)(size_t)&Bs4[tid];
  const int bn  = n0 + (tid >> 2);                  // B row (column n) this thread stages
  const int bc4 = tid & 3;

  v8f zero = {0.f,0.f,0.f,0.f,0.f,0.f,0.f,0.f};
  v8f c[2][2];
  c[0][0] = zero; c[0][1] = zero; c[1][0] = zero; c[1][1] = zero;

  const int ml   = lane & 15;
  const int kb2  = (lane >> 4) << 2;   // A: kbase/2 in dwords (0 or 4)
  const int kof2 = (lane >> 4) << 3;   // B: kbase/2 in dwords (0 or 8)

  for (int k0 = 0; k0 < Kpad; k0 += BK) {
    // A tile: 128x32 bf16 = 512 uint4, async copied (2 per thread)
    async_copy16(asOff0, (const void*)(A32 + aOff0 + (k0 >> 1)));
    async_copy16(asOff1, (const void*)(A32 + aOff1 + (k0 >> 1)));
    // B tile: one packed panel slice, 64 rows x 4 uint4 (1 per thread)
    long long bIdx = ((long long)(k0 >> 5) * Npad + bn) * 4 + bc4;
    if (fullB) {
      async_copy16(bsOff, (const void*)(B4 + bIdx));
    } else {
      uint4 bv = {0u, 0u, 0u, 0u};
      if (bn < Npad) bv = B4[bIdx];
      Bs4[tid] = bv;
    }
    async_wait0();
    __syncthreads();

    Frag a[2], b[2];
    #pragma unroll
    for (int tm = 0; tm < 2; ++tm) {
      int m = waveM * 32 + tm * 16 + ml;
      #pragma unroll
      for (int j = 0; j < 4; ++j) {
        a[tm].d[j]     = As32[m * 16 + kb2 + j];        // K = kb..kb+7
        a[tm].d[4 + j] = As32[m * 16 + kb2 + 8 + j];    // K = kb+16..kb+23
      }
    }
    #pragma unroll
    for (int tn = 0; tn < 2; ++tn) {
      int n = waveN * 32 + tn * 16 + ml;
      #pragma unroll
      for (int j = 0; j < 4; ++j) {
        b[tn].d[j]     = Bs32[n * 16 + kb2 + j];        // K = kb..kb+7
        b[tn].d[4 + j] = Bs32[n * 16 + kb2 + 8 + j];    // K = kb+16..kb+23
      }
    }
    #pragma unroll
    for (int tm = 0; tm < 2; ++tm)
      #pragma unroll
      for (int tn = 0; tn < 2; ++tn)
        c[tm][tn] = __builtin_amdgcn_wmma_f32_16x16x32_bf16(
            false, a[tm].v, false, b[tn].v, (short)0, c[tm][tn], false, false);
    __syncthreads();
  }

  // epilogue: bias + optional selu, dual f32/bf16 stores
  const int mo = (lane >> 4) << 3;
  #pragma unroll
  for (int tm = 0; tm < 2; ++tm) {
    #pragma unroll
    for (int tn = 0; tn < 2; ++tn) {
      int nc = n0 + waveN * 32 + tn * 16 + ml;
      if (nc >= Npad) continue;
      float bv = (bias != nullptr && nc < Nreal) ? bias[nc] : 0.f;
      int mb = m0 + waveM * 32 + tm * 16 + mo;
      #pragma unroll
      for (int i = 0; i < 8; ++i) {
        float v = c[tm][tn][i] + bv;
        if (act) v = selu_f(v);
        long long row = (long long)bz * M + mb + i;
        if (Cf) Cf[row * ldc  + nc] = v;
        if (Cb) Cb[row * ldcb + nc] = f2bf(v);
      }
    }
  }
}

// -------------------------------------- fused pairwise-distance adjacency
// adj[b][i][j] = exp(-sqrt(clip(na_i - 2*dc_i.dc_j + na_j, 1e-12, 1e12)))
// dot via one WMMA K-step (DIST_DIM = 32); adjacency stored bf16 (L2-resident).
__global__ __launch_bounds__(256) void k_adj(
    const unsigned short* __restrict__ dc,   // [2*N x 32] bf16
    const float* __restrict__ na,            // [2*N]
    unsigned short* __restrict__ adj,        // [2][N][N] bf16
    int N)
{
  const int b  = blockIdx.z;
  const int i0 = blockIdx.y * 128;
  const int j0 = blockIdx.x * 64;
  const int tid  = threadIdx.x;
  const int lane = tid & 31;
  const int wave = tid >> 5;
  const int waveM = wave >> 1, waveN = wave & 1;

  __shared__ uint4 Ai4[512];   // 128 rows x 4 uint4 (32 bf16 each)
  __shared__ uint4 Bj4[256];   // 64 rows x 4 uint4
  unsigned int* Ai32 = (unsigned int*)Ai4;
  unsigned int* Bj32 = (unsigned int*)Bj4;
  const uint4* dc4 = (const uint4*)dc;
  const long long rbase = (long long)b * N;

  #pragma unroll
  for (int i = 0; i < 2; ++i) {
    int u = tid + i * 256;
    Ai4[u] = dc4[(rbase + i0 + (u >> 2)) * 4 + (u & 3)];
  }
  Bj4[tid] = dc4[(rbase + j0 + (tid >> 2)) * 4 + (tid & 3)];
  __syncthreads();

  const int ml  = lane & 15;
  const int kb2 = (lane >> 4) << 2;   // A kbase/2 (dwords)
  const int kb3 = (lane >> 4) << 3;   // B: dc_j row is already K-contiguous

  Frag a[2], bb[2];
  #pragma unroll
  for (int tm = 0; tm < 2; ++tm) {
    int m = waveM * 32 + tm * 16 + ml;
    #pragma unroll
    for (int j = 0; j < 4; ++j) {
      a[tm].d[j]     = Ai32[m * 16 + kb2 + j];
      a[tm].d[4 + j] = Ai32[m * 16 + kb2 + 8 + j];
    }
  }
  #pragma unroll
  for (int tn = 0; tn < 2; ++tn) {
    int n = waveN * 32 + tn * 16 + ml;
    #pragma unroll
    for (int j = 0; j < 4; ++j) {
      bb[tn].d[j]     = Bj32[n * 16 + kb2 + j];
      bb[tn].d[4 + j] = Bj32[n * 16 + kb2 + 8 + j];
    }
  }

  v8f zero = {0.f,0.f,0.f,0.f,0.f,0.f,0.f,0.f};
  v8f c[2][2];
  #pragma unroll
  for (int tm = 0; tm < 2; ++tm)
    #pragma unroll
    for (int tn = 0; tn < 2; ++tn)
      c[tm][tn] = __builtin_amdgcn_wmma_f32_16x16x32_bf16(
          false, a[tm].v, false, bb[tn].v, (short)0, zero, false, false);

  const int mo = (lane >> 4) << 3;
  #pragma unroll
  for (int tm = 0; tm < 2; ++tm) {
    #pragma unroll
    for (int tn = 0; tn < 2; ++tn) {
      int j = j0 + waveN * 32 + tn * 16 + ml;
      float nb = na[rbase + j];
      int ib = i0 + waveM * 32 + tm * 16 + mo;
      #pragma unroll
      for (int i = 0; i < 8; ++i) {
        int irow = ib + i;
        float dsq = na[rbase + irow] - 2.f * c[tm][tn][i] + nb;
        dsq = fminf(fmaxf(dsq, 1e-12f), 1e12f);
        adj[(rbase + irow) * (long long)N + j] = f2bf(expf(-sqrtf(dsq)));
      }
    }
  }
}

// ---------------------------------------------------- pointwise kernels
// Weights converted to bf16 directly in packed-B layout.
__global__ void k_padw(const float* __restrict__ W, unsigned short* __restrict__ Wp,
                       int K, int N, int Kpad, int Npad) {
  int t = blockIdx.x * 256 + threadIdx.x;
  if (t >= Kpad * Npad) return;
  int k = t / Npad, n = t % Npad;
  float v = (k < K && n < N) ? W[(long long)k * N + n] : 0.f;
  int p = k >> 5, kk = k & 31;
  long long idx = ((long long)(p * Npad + n) * 16 + (kk >> 1)) * 2 + (kk & 1);
  Wp[idx] = f2bf(v);
}

__global__ void k_encode(const float* __restrict__ X, unsigned short* __restrict__ enc, int M) {
  int m = blockIdx.x * 256 + threadIdx.x;
  if (m >= M) return;
  const float* x = X + (long long)m * 15;
  int id = (int)x[0];
  unsigned short* e = enc + (long long)m * 32;
  #pragma unroll
  for (int c = 0; c < 32; ++c) {
    if (c >= 25) { e[c] = 0; continue; }
    float v = (c < 11) ? ((c == id) ? 1.f : 0.f) : x[c - 10];
    e[c] = f2bf(selu_f(v));
  }
}

__global__ void k_na(const unsigned short* __restrict__ dc, float* __restrict__ na, int M) {
  int m = blockIdx.x * 256 + threadIdx.x;
  if (m >= M) return;
  const unsigned short* p = dc + (long long)m * 32;
  float s = 0.f;
  #pragma unroll
  for (int c = 0; c < 32; ++c) { float v = bf2f(p[c]); s += v * v; }
  na[m] = s;
}

__global__ __launch_bounds__(256) void k_rownorm(const unsigned short* __restrict__ adj,
                                                 float* __restrict__ nrm, int N) {
  long long row = blockIdx.x;
  const uint4* p4 = (const uint4*)(adj + row * N);
  float s = 0.f;
  for (int i = threadIdx.x; i < (N >> 3); i += 256) {
    uint4 v = p4[i];
    unsigned int d[4] = {v.x, v.y, v.z, v.w};
    #pragma unroll
    for (int q = 0; q < 4; ++q) {
      s += bf2f((unsigned short)(d[q] & 0xFFFFu));
      s += bf2f((unsigned short)(d[q] >> 16));
    }
  }
  __shared__ float red[256];
  red[threadIdx.x] = s;
  __syncthreads();
  for (int st = 128; st > 0; st >>= 1) {
    if (threadIdx.x < st) red[threadIdx.x] += red[threadIdx.x + st];
    __syncthreads();
  }
  if (threadIdx.x == 0) nrm[row] = rsqrtf(red[0] + 1e-6f);   // (deg+1e-6)^-0.5
}

// support*norm -> bf16 in packed-B layout (per batch, Npad=256)
__global__ void k_scale(const float* __restrict__ sup, const float* __restrict__ nrm,
                        unsigned short* __restrict__ s2, int N, long long total) {
  long long t = (long long)blockIdx.x * 256 + threadIdx.x;
  if (t >= total) return;
  long long grow = t >> 8;
  int n = (int)(t & 255);
  int b = (int)(grow / N);
  int k = (int)(grow - (long long)b * N);
  int p = k >> 5, kk = k & 31;
  long long idx = (long long)b * N * 256 +
                  ((long long)(p * 256 + n) * 16 + (kk >> 1)) * 2 + (kk & 1);
  s2[idx] = f2bf(sup[t] * nrm[grow]);
}

__global__ void k_convpost(const float* __restrict__ cf, const float* __restrict__ nrm,
                           const float* __restrict__ bias, unsigned short* __restrict__ xb,
                           long long total) {
  long long t = (long long)blockIdx.x * 256 + threadIdx.x;
  if (t >= total) return;
  xb[t] = f2bf(selu_f(cf[t] * nrm[t >> 8] + bias[(int)(t & 255)]));
}

__global__ void k_mcat(const unsigned short* __restrict__ x2, const float* __restrict__ logits,
                       unsigned short* __restrict__ mc, int M) {
  long long t = (long long)blockIdx.x * 256 + threadIdx.x;
  if (t >= (long long)M * 288) return;
  int row = (int)(t / 288), c = (int)(t % 288);
  unsigned short v;
  if (c < 256)      v = x2[(long long)row * 256 + c];
  else if (c < 264) v = f2bf(selu_f(logits[(long long)row * 16 + (c - 256)]));
  else              v = 0;
  mc[t] = v;
}

__global__ void k_final(const float* __restrict__ logits, const float* __restrict__ mom,
                        const float* __restrict__ charge, float* __restrict__ out, int M) {
  int m = blockIdx.x * 256 + threadIdx.x;
  if (m >= M) return;
  const float* l = logits + (long long)m * 16;
  int arg = 0; float best = l[0];
  #pragma unroll
  for (int k = 1; k < 8; ++k) if (l[k] > best) { best = l[k]; arg = k; }
  float msk = (arg != 0) ? 1.f : 0.f;
  float* o = out + (long long)m * 12;
  #pragma unroll
  for (int k = 0; k < 8; ++k) o[k] = l[k];
  #pragma unroll
  for (int k = 0; k < 3; ++k) o[8 + k] = mom[(long long)m * 16 + k] * msk;
  o[11] = charge[(long long)m * 16];
}

// ------------------------------------------------------------- host side
extern "C" void kernel_launch(void* const* d_in, const int* in_sizes, int n_in,
                              void* d_out, int out_size, void* d_ws, size_t ws_size,
                              hipStream_t stream) {
  (void)in_sizes; (void)n_in; (void)out_size; (void)ws_size;
  const int Bn = 2, N = 6400, M = Bn * N;
  const float* X = (const float*)d_in[0];
  auto Wf = [&](int i) { return (const float*)d_in[1 + 2 * i]; };
  auto Bf = [&](int i) { return (const float*)d_in[2 + 2 * i]; };

  size_t off = 0;
  auto alloc = [&](size_t bytes) -> void* {
    off = (off + 255) & ~(size_t)255;
    void* p = (char*)d_ws + off;
    off += bytes;
    return p;
  };

  struct LM { int K, N, Kp, Np; };
  // order matches _LAYERS
  static const LM L[20] = {
    {25,256,32,256},{256,256,256,256},{256,256,256,256},{256,32,256,32},   // distcoords1..3, distcoords
    {25,256,32,256},{256,256,256,256},{256,256,256,256},                   // input1..3
    {256,256,256,256},{256,256,256,256},                                   // conv1, conv2
    {256,256,256,256},{256,256,256,256},{256,256,256,256},{256,256,256,256}, // id1..4
    {256,8,256,16},{256,1,256,16},                                         // out_id, out_charge
    {264,264,288,288},{264,256,288,256},                                   // momentum1, momentum2
    {256,256,256,256},{256,256,256,256},                                   // momentum3, momentum4
    {256,3,256,16}                                                         // out_momentum
  };
  unsigned short* wp[20];
  for (int i = 0; i < 20; ++i)
    wp[i] = (unsigned short*)alloc((size_t)L[i].Kp * L[i].Np * 2);

  unsigned short* enc  = (unsigned short*)alloc((size_t)M * 32 * 2);
  unsigned short* t0   = (unsigned short*)alloc((size_t)M * 288 * 2);
  unsigned short* t1   = (unsigned short*)alloc((size_t)M * 288 * 2);
  unsigned short* dcb  = (unsigned short*)alloc((size_t)M * 32 * 2);
  unsigned short* xb   = (unsigned short*)alloc((size_t)M * 256 * 2);
  unsigned short* s2   = (unsigned short*)alloc((size_t)M * 256 * 2);   // packed-B layout
  unsigned short* x1b  = (unsigned short*)alloc((size_t)M * 256 * 2);
  unsigned short* x2b  = (unsigned short*)alloc((size_t)M * 256 * 2);
  unsigned short* mcat = (unsigned short*)alloc((size_t)M * 288 * 2);
  unsigned short* adj  = (unsigned short*)alloc((size_t)Bn * N * N * 2); // 164 MB bf16
  float* supf    = (float*)alloc((size_t)M * 256 * 4);
  float* convf   = (float*)alloc((size_t)M * 256 * 4);
  float* logits  = (float*)alloc((size_t)M * 16 * 4);
  float* chargef = (float*)alloc((size_t)M * 16 * 4);
  float* momf    = (float*)alloc((size_t)M * 16 * 4);
  float* naf     = (float*)alloc((size_t)M * 4);
  float* nrmf    = (float*)alloc((size_t)M * 4);

  for (int i = 0; i < 20; ++i) {
    int total = L[i].Kp * L[i].Np;
    k_padw<<<(total + 255) / 256, 256, 0, stream>>>(Wf(i), wp[i], L[i].K, L[i].N, L[i].Kp, L[i].Np);
  }
  k_encode<<<(M + 255) / 256, 256, 0, stream>>>(X, enc, M);

  auto gemm = [&](const unsigned short* A, int lda, long long sA,
                  const unsigned short* W, long long sB,
                  const float* bias, float* Cf, int ldc,
                  unsigned short* Cb, int ldcb,
                  int Mm, int Np, int Kp, int Nr, int act, int batches) {
    dim3 g((Np + BN - 1) / BN, Mm / BM, batches);
    k_gemm<<<g, 256, 0, stream>>>(A, lda, sA, W, sB, bias, Cf, ldc, Cb, ldcb,
                                  Mm, Np, Kp, Nr, act);
  };

  const long long MH   = (long long)M * 256;
  const long long M288 = (long long)M * 288;

  // distance-coordinate MLP -> adjacency
  gemm(enc, 32, 0, wp[0], 0, Bf(0), nullptr, 0, t0, 256, M, 256, 32, 256, 1, 1);
  gemm(t0, 256, 0, wp[1], 0, Bf(1), nullptr, 0, t1, 256, M, 256, 256, 256, 1, 1);
  gemm(t1, 256, 0, wp[2], 0, Bf(2), nullptr, 0, t0, 256, M, 256, 256, 256, 1, 1);
  gemm(t0, 256, 0, wp[3], 0, Bf(3), nullptr, 0, dcb, 32, M, 32, 256, 32, 0, 1);
  k_na<<<(M + 255) / 256, 256, 0, stream>>>(dcb, naf, M);
  {
    dim3 g(N / 64, N / 128, Bn);
    k_adj<<<g, 256, 0, stream>>>(dcb, naf, adj, N);
  }
  k_rownorm<<<M, 256, 0, stream>>>(adj, nrmf, N);

  // node-feature MLP
  gemm(enc, 32, 0, wp[4], 0, Bf(4), nullptr, 0, t0, 256, M, 256, 32, 256, 1, 1);
  gemm(t0, 256, 0, wp[5], 0, Bf(5), nullptr, 0, t1, 256, M, 256, 256, 256, 1, 1);
  gemm(t1, 256, 0, wp[6], 0, Bf(6), nullptr, 0, xb, 256, M, 256, 256, 256, 1, 1);

  // graph conv 1:  selu( (adj @ (x@W * norm)) * norm + b )
  gemm(xb, 256, 0, wp[7], 0, nullptr, supf, 256, nullptr, 0, M, 256, 256, 256, 0, 1);
  k_scale<<<(unsigned)((MH + 255) / 256), 256, 0, stream>>>(supf, nrmf, s2, N, MH);
  gemm(adj, N, (long long)N * N, s2, (long long)N * 256, nullptr,
       convf, 256, nullptr, 0, N, 256, N, 256, 0, Bn);
  k_convpost<<<(unsigned)((MH + 255) / 256), 256, 0, stream>>>(convf, nrmf, Bf(7), x1b, MH);

  // id / charge head
  gemm(x1b, 256, 0, wp[9], 0, Bf(9), nullptr, 0, t0, 256, M, 256, 256, 256, 1, 1);
  gemm(t0, 256, 0, wp[10], 0, Bf(10), nullptr, 0, t1, 256, M, 256, 256, 256, 1, 1);
  gemm(t1, 256, 0, wp[11], 0, Bf(11), nullptr, 0, t0, 256, M, 256, 256, 256, 1, 1);
  gemm(t0, 256, 0, wp[12], 0, Bf(12), nullptr, 0, t1, 256, M, 256, 256, 256, 1, 1);
  gemm(t1, 256, 0, wp[13], 0, Bf(13), logits, 16, nullptr, 0, M, 16, 256, 8, 0, 1);
  gemm(t1, 256, 0, wp[14], 0, Bf(14), chargef, 16, nullptr, 0, M, 16, 256, 1, 0, 1);

  // graph conv 2
  gemm(xb, 256, 0, wp[8], 0, nullptr, supf, 256, nullptr, 0, M, 256, 256, 256, 0, 1);
  k_scale<<<(unsigned)((MH + 255) / 256), 256, 0, stream>>>(supf, nrmf, s2, N, MH);
  gemm(adj, N, (long long)N * N, s2, (long long)N * 256, nullptr,
       convf, 256, nullptr, 0, N, 256, N, 256, 0, Bn);
  k_convpost<<<(unsigned)((MH + 255) / 256), 256, 0, stream>>>(convf, nrmf, Bf(8), x2b, MH);

  // momentum head
  k_mcat<<<(unsigned)((M288 + 255) / 256), 256, 0, stream>>>(x2b, logits, mcat, M);
  gemm(mcat, 288, 0, wp[15], 0, Bf(15), nullptr, 0, t0, 288, M, 288, 288, 264, 1, 1);
  gemm(t0, 288, 0, wp[16], 0, Bf(16), nullptr, 0, t1, 256, M, 256, 288, 256, 1, 1);
  gemm(t1, 256, 0, wp[17], 0, Bf(17), nullptr, 0, t0, 256, M, 256, 256, 256, 1, 1);
  gemm(t0, 256, 0, wp[18], 0, Bf(18), nullptr, 0, t1, 256, M, 256, 256, 256, 1, 1);
  gemm(t1, 256, 0, wp[19], 0, Bf(19), momf, 16, nullptr, 0, M, 16, 256, 3, 0, 1);

  k_final<<<(M + 255) / 256, 256, 0, stream>>>(logits, momf, chargef, (float*)d_out, M);
}
